// CrossAttnFusion_16595753632188
// MI455X (gfx1250) — matrix operand order repaired and checked by reading, not compiled, not run on hardware
//
#include <hip/hip_runtime.h>
#include <hip/hip_bf16.h>
#include <math.h>

#define D 256
#define B_ROWS 65536
#define RPB 64           // rows per block
#define EPS 1e-5f

typedef __attribute__((ext_vector_type(16))) __bf16 bf16x16;
typedef __attribute__((ext_vector_type(8)))  float  f32x8;

union Frag {
  bf16x16 v;
  uint4   q[2];
};

__device__ __forceinline__ unsigned short f2bf(float f) {
  unsigned int u = __float_as_uint(f);
  unsigned int r = ((u >> 16) & 1u) + 0x7fffu;   // round-to-nearest-even
  return (unsigned short)((u + r) >> 16);
}

// A-matrix 16x32 bf16 fragment from LDS (row-major, stride in halfwords).
// Lanes 0-15: M=0..15, K={0..7,16..23}; lanes 16-31: M=0..15, K={8..15,24..31}.
__device__ __forceinline__ Frag load_a_lds(const unsigned short* base, int stride,
                                           int m0, int k0, int lane) {
  Frag f;
  const int m  = m0 + (lane & 15);
  const int bk = (lane < 16) ? 0 : 8;
  const unsigned short* p = base + m * stride + k0 + bk;
  f.q[0] = *(const uint4*)(p);        // K = bk .. bk+7
  f.q[1] = *(const uint4*)(p + 16);   // K = 16+bk .. 16+bk+7
  return f;
}

// B-matrix 32x16 bf16 fragment B[k][n] = W[n][k], W row-major (ld halfwords).
// Lane L: n = n0 + L%16, K = k0 + {0..7,16..23} (L<16) or {8..15,24..31} (L>=16).
__device__ __forceinline__ Frag load_b_glb(const unsigned short* W, int ld,
                                           int n0, int k0, int lane) {
  Frag f;
  const int n  = n0 + (lane & 15);
  const int kk = k0 + ((lane < 16) ? 0 : 8);
  const unsigned short* p = W + (size_t)n * ld + kk;
  f.q[0] = *(const uint4*)(p);
  f.q[1] = *(const uint4*)(p + 16);
  return f;
}

__device__ __forceinline__ f32x8 wmma_bf16(bf16x16 a, bf16x16 b, f32x8 c) {
  return __builtin_amdgcn_wmma_f32_16x16x32_bf16(false, a, false, b,
                                                 (short)0, c, false, false);
}

// ---------------- prep: fold W_comb = out_w @ Wv, b_comb; cast w1,w2 -> bf16
__global__ void prep_kernel(const float* __restrict__ in_w, const float* __restrict__ in_b,
                            const float* __restrict__ out_w, const float* __restrict__ out_b,
                            const float* __restrict__ w1, const float* __restrict__ w2,
                            unsigned short* __restrict__ Wc, unsigned short* __restrict__ w1b,
                            unsigned short* __restrict__ w2b, float* __restrict__ bcomb) {
  const int bid = blockIdx.x, t = threadIdx.x;
  if (bid < 256) {                        // W_comb[n][k] = sum_j out_w[n][j]*in_w[2D+j][k]
    const int n = bid, k = t;
    const float* ow = out_w + n * D;
    float s = 0.f;
    for (int j = 0; j < D; ++j) s += ow[j] * in_w[(size_t)(2 * D + j) * D + k];
    Wc[n * D + k] = f2bf(s);
  } else if (bid < 256 + 1024) {          // w1 cast (1024 x 256)
    const int r = bid - 256;
    w1b[r * D + t] = f2bf(w1[r * D + t]);
  } else if (bid < 256 + 1024 + 256) {    // w2 cast (256 x 1024)
    const int r = bid - 1280;
    for (int i = 0; i < 4; ++i)
      w2b[(size_t)r * 1024 + t + 256 * i] = f2bf(w2[(size_t)r * 1024 + t + 256 * i]);
  } else {                                // b_comb = out_w @ bv + out_b
    const float* ow = out_w + t * D;
    float s = out_b[t];
    for (int j = 0; j < D; ++j) s += ow[j] * in_b[2 * D + j];
    bcomb[t] = s;
  }
}

// ---------------- fully fused:
//   h   = e_raw + LN(e_feat)@Wc^T + bcomb      (h stays in VGPR accumulators)
//   lnh = LN(h)                                 (bf16, written back into LDS A)
//   out = h + gelu(lnh@w1^T+b1)@w2^T + b2       (ff1 chunked through LDS)
__global__ void __launch_bounds__(256)
fused_kernel(const float* __restrict__ e_raw, const float* __restrict__ e_feat,
             const float* __restrict__ kvn_g, const float* __restrict__ kvn_b,
             const float* __restrict__ ffn_g, const float* __restrict__ ffn_b,
             const unsigned short* __restrict__ Wc, const float* __restrict__ bcomb,
             const unsigned short* __restrict__ w1b, const float* __restrict__ b1,
             const unsigned short* __restrict__ w2b, const float* __restrict__ b2,
             float* __restrict__ out) {
  __shared__ unsigned short A[RPB][D + 8];       // e_feat-LN tile, later lnh tile
  __shared__ unsigned short Cc[RPB][128 + 8];    // gelu(ff1) chunk
  __shared__ float sums[RPB][2];
  __shared__ float sqs[RPB][2];
  __shared__ float meanv[RPB];
  __shared__ float rstdv[RPB];

  const int t = threadIdx.x;
  const int lane = t & 31;
  const int w = t >> 5;
  const long rowBase = (long)blockIdx.x * RPB;

  // ---- Phase 1: LN(e_feat) -> A (bf16), 4 lanes per row
  {
    const int row = t >> 2;
    const int q = t & 3;
    const float* src = e_feat + (rowBase + row) * D + q * 64;
    float x[64];
    float s = 0.f, ss = 0.f;
#pragma unroll
    for (int i = 0; i < 16; ++i) {
      float4 v = ((const float4*)src)[i];
      x[4*i+0]=v.x; x[4*i+1]=v.y; x[4*i+2]=v.z; x[4*i+3]=v.w;
      s  += v.x + v.y + v.z + v.w;
      ss += v.x*v.x + v.y*v.y + v.z*v.z + v.w*v.w;
    }
    s  += __shfl_xor(s, 1, 32);  s  += __shfl_xor(s, 2, 32);
    ss += __shfl_xor(ss, 1, 32); ss += __shfl_xor(ss, 2, 32);
    const float mean = s * (1.f / D);
    const float var  = ss * (1.f / D) - mean * mean;
    const float inv  = rsqrtf(var + EPS);
#pragma unroll
    for (int i = 0; i < 64; ++i) {
      const int k = q * 64 + i;
      A[row][k] = f2bf((x[i] - mean) * inv * kvn_g[k] + kvn_b[k]);
    }
  }
  __syncthreads();

  const int mt = w >> 1, half = w & 1, m0 = 16 * mt;
  const int rOff = (lane < 16) ? 0 : 8;
  const int nLane = lane & 15;

  // ---- Phase 2: register-resident A fragments (each wave owns a 16-row stripe)
  f32x8 accs[8];   // becomes h
  {
    Frag Areg[8];
#pragma unroll
    for (int kf = 0; kf < 8; ++kf)
      Areg[kf] = load_a_lds(&A[0][0], D + 8, m0, 32 * kf, lane);

    // ---- Phase 3: GEMM (attention path folded into one 256x256 GEMM)
#pragma unroll
    for (int j = 0; j < 8; ++j) {
      const int n0 = 16 * (8 * half + j);
      f32x8 acc = {0.f,0.f,0.f,0.f,0.f,0.f,0.f,0.f};
#pragma unroll
      for (int kf = 0; kf < 8; ++kf) {
        Frag b = load_b_glb(Wc, D, n0, 32 * kf, lane);
        acc = wmma_bf16(Areg[kf].v, b.v, acc);
      }
      accs[j] = acc;
    }
  }

  // ---- Epilogue: h = acc + bcomb + e_raw (kept in regs); gather row stats
  float ps[8], pss[8];
#pragma unroll
  for (int r = 0; r < 8; ++r) { ps[r] = 0.f; pss[r] = 0.f; }
#pragma unroll
  for (int j = 0; j < 8; ++j) {
    const int col = 16 * (8 * half + j) + nLane;
    const float bc = bcomb[col];
#pragma unroll
    for (int r = 0; r < 8; ++r) {
      const int rowL = m0 + r + rOff;
      const long gi = (rowBase + rowL) * D + col;
      const float hv = accs[j][r] + bc + e_raw[gi];
      accs[j][r] = hv;
      ps[r]  += hv;
      pss[r] += hv * hv;
    }
  }
#pragma unroll
  for (int r = 0; r < 8; ++r) {
#pragma unroll
    for (int m = 1; m <= 8; m <<= 1) {   // reduce within 16-lane group
      ps[r]  += __shfl_xor(ps[r],  m, 32);
      pss[r] += __shfl_xor(pss[r], m, 32);
    }
  }
  if (nLane == 0) {
#pragma unroll
    for (int r = 0; r < 8; ++r) {
      const int rowL = m0 + r + rOff;
      sums[rowL][half] = ps[r];
      sqs[rowL][half]  = pss[r];
    }
  }
  __syncthreads();   // also guarantees every wave is done reading A (phase 2)

  if (t < RPB) {
    const float s  = sums[t][0] + sums[t][1];
    const float ss = sqs[t][0] + sqs[t][1];
    const float mean = s * (1.f / D);
    const float var  = ss * (1.f / D) - mean * mean;
    meanv[t] = mean;
    rstdv[t] = rsqrtf(var + EPS);
  }
  __syncthreads();

  // ---- Phase 4: lnh = LN(h) in bf16, written back into LDS A (overwrite)
#pragma unroll
  for (int j = 0; j < 8; ++j) {
    const int col = 16 * (8 * half + j) + nLane;
    const float g = ffn_g[col], bb = ffn_b[col];
#pragma unroll
    for (int r = 0; r < 8; ++r) {
      const int rowL = m0 + r + rOff;
      A[rowL][col] = f2bf((accs[j][r] - meanv[rowL]) * rstdv[rowL] * g + bb);
    }
  }
  __syncthreads();

  // ---- Phase 5: FFN. A2 fragments of lnh, then K-chunked gelu+GEMM2.
  Frag Areg[8];
#pragma unroll
  for (int kf = 0; kf < 8; ++kf)
    Areg[kf] = load_a_lds(&A[0][0], D + 8, m0, 32 * kf, lane);

  const f32x8 zero = {0.f,0.f,0.f,0.f,0.f,0.f,0.f,0.f};
  f32x8 acc2[8];
#pragma unroll
  for (int j = 0; j < 8; ++j) acc2[j] = zero;

  for (int c = 0; c < 8; ++c) {          // 8 chunks of 128 ff1 columns
    const int cb = 128 * c;
    // GEMM1 chunk: ff1[64 x 128] = lnh @ w1[cb:cb+128]^T, +b1, gelu -> Cc bf16
#pragma unroll
    for (int j = 0; j < 4; ++j) {
      const int n0c = 16 * (4 * half + j);
      f32x8 acc = zero;
#pragma unroll
      for (int kf = 0; kf < 8; ++kf) {
        Frag b = load_b_glb(w1b, D, cb + n0c, 32 * kf, lane);
        acc = wmma_bf16(Areg[kf].v, b.v, acc);
      }
      const int colc = n0c + nLane;
      const float bb = b1[cb + colc];
#pragma unroll
      for (int r = 0; r < 8; ++r) {
        const int rowL = m0 + r + rOff;
        const float xg = acc[r] + bb;
        const float ge = 0.5f * xg * (1.f + erff(xg * 0.70710678118f)); // exact gelu
        Cc[rowL][colc] = f2bf(ge);
      }
    }
    __syncthreads();
    // GEMM2 partial: acc2 += Cc(64x128) @ w2[:, cb:cb+128]^T
#pragma unroll
    for (int kf2 = 0; kf2 < 4; ++kf2) {
      Frag a2 = load_a_lds(&Cc[0][0], 128 + 8, m0, 32 * kf2, lane);
#pragma unroll
      for (int j = 0; j < 8; ++j) {
        const int n0 = 16 * (8 * half + j);
        Frag b = load_b_glb(w2b, 4 * D, n0, cb + 32 * kf2, lane);
        acc2[j] = wmma_bf16(a2.v, b.v, acc2[j]);
      }
    }
    __syncthreads();
  }

  // ---- Final epilogue: out = h(reg) + ff2 + b2 — single HBM write
#pragma unroll
  for (int j = 0; j < 8; ++j) {
    const int col = 16 * (8 * half + j) + nLane;
    const float bb = b2[col];
#pragma unroll
    for (int r = 0; r < 8; ++r) {
      const int rowL = m0 + r + rOff;
      const long gi = (rowBase + rowL) * D + col;
      out[gi] = accs[j][r] + acc2[j][r] + bb;
    }
  }
}

extern "C" void kernel_launch(void* const* d_in, const int* in_sizes, int n_in,
                              void* d_out, int out_size, void* d_ws, size_t ws_size,
                              hipStream_t stream) {
  const float* e_raw  = (const float*)d_in[0];
  const float* e_feat = (const float*)d_in[1];
  const float* kvn_g  = (const float*)d_in[4];
  const float* kvn_b  = (const float*)d_in[5];
  const float* in_w   = (const float*)d_in[6];
  const float* in_b   = (const float*)d_in[7];
  const float* out_w  = (const float*)d_in[8];
  const float* out_b  = (const float*)d_in[9];
  const float* ffn_g  = (const float*)d_in[10];
  const float* ffn_b  = (const float*)d_in[11];
  const float* w1     = (const float*)d_in[12];
  const float* b1     = (const float*)d_in[13];
  const float* w2     = (const float*)d_in[14];
  const float* b2     = (const float*)d_in[15];

  unsigned char* ws = (unsigned char*)d_ws;
  unsigned short* Wc  = (unsigned short*)(ws);                              // 128KB
  unsigned short* w1b = (unsigned short*)(ws + 131072);                     // 512KB
  unsigned short* w2b = (unsigned short*)(ws + 131072 + 524288);            // 512KB
  float*          bcb = (float*)         (ws + 131072 + 524288 + 524288);   // 1KB

  prep_kernel<<<1537, 256, 0, stream>>>(in_w, in_b, out_w, out_b, w1, w2,
                                        Wc, w1b, w2b, bcb);
  fused_kernel<<<B_ROWS / RPB, 256, 0, stream>>>(e_raw, e_feat, kvn_g, kvn_b,
                                                 ffn_g, ffn_b, Wc, bcb,
                                                 w1b, b1, w2b, b2,
                                                 (float*)d_out);
}
